// CNNModel_42064909697048
// MI455X (gfx1250) — matrix operand order repaired and checked by reading, not compiled, not run on hardware
//
#include <hip/hip_runtime.h>

typedef _Float16 v8h  __attribute__((ext_vector_type(8)));
typedef _Float16 v16h __attribute__((ext_vector_type(16)));
typedef float    v8f  __attribute__((ext_vector_type(8)));

#define TILE_M 64
#define FSTR   424   // 416 K-pad + 8 LDS pad
#define A1STR  296   // 288 + 8
#define A2STR  104   // 96 + 8
#define A3STR  40    // 32 + 8

// packed-weight offsets (in _Float16 elements) inside d_ws
#define OFF_W1 0
#define OFF_W2 (13*17*512)              // 113152
#define OFF_W3 (OFF_W2 + 9*6*512)       // 140800
#define OFF_W4 (OFF_W3 + 3*2*512)       // 143872
#define W_END  (OFF_W4 + 1*1*512)       // 144384 halves = 288768 bytes
#define BIAS_BYTE_OFF (W_END*2)

static __device__ __forceinline__ unsigned f2key(float x) {
  unsigned b = __float_as_uint(x);
  return (b & 0x80000000u) ? ~b : (b | 0x80000000u);
}
static __device__ __forceinline__ float key2f(unsigned u) {
  return __uint_as_float((u & 0x80000000u) ? (u ^ 0x80000000u) : ~u);
}

// One reduction task over 28 elements (a row or a column of one image).
// min/argmin/max/argmax/mean + stable lower-median value & index.
// __noinline__ keeps exactly one copy of the unrolled body in the binary.
static __device__ __noinline__ void stat_task(
    const float* p, int stride, _Float16* dst,
    int o_minv, int o_mini, int o_maxv, int o_maxi,
    int o_mean, int o_medv, int o_medi)
{
  unsigned u[28];
  int      rank[28];
  float x0 = p[0];
  float mn = x0, mx = x0, s = x0;
  int mni = 0, mxi = 0;
  u[0] = f2key(x0); rank[0] = 0;
#pragma unroll
  for (int j = 1; j < 28; ++j) {
    float x = p[j * stride];
    s += x;
    if (x < mn) { mn = x; mni = j; }   // first occurrence == jnp.argmin
    if (x > mx) { mx = x; mxi = j; }   // first occurrence == jnp.argmax
    u[j] = f2key(x); rank[j] = 0;
  }
  // stable ranks: for i<j, j precedes i iff u[j] < u[i] (ties -> i first)
#pragma unroll
  for (int i = 0; i < 27; ++i) {
#pragma unroll
    for (int j = i + 1; j < 28; ++j) {
      int c = (u[j] < u[i]) ? 1 : 0;
      rank[i] += c;
      rank[j] += 1 - c;
    }
  }
  unsigned mu = 0; int mi = 0;
#pragma unroll
  for (int j = 0; j < 28; ++j) {
    int hit = (rank[j] == 13) ? 1 : 0;   // lower median: 13 elements before it
    mu = hit ? u[j] : mu;
    mi = hit ? j    : mi;
  }
  dst[o_minv] = (_Float16)mn;
  dst[o_mini] = (_Float16)(float)mni;
  dst[o_maxv] = (_Float16)mx;
  dst[o_maxi] = (_Float16)(float)mxi;
  dst[o_mean] = (_Float16)(s * (1.0f / 28.0f));
  dst[o_medv] = (_Float16)key2f(mu);
  dst[o_medi] = (_Float16)(float)mi;
}

static __device__ __forceinline__ v16h cat16(v8h a, v8h b) {
  v16h r;
#pragma unroll
  for (int i = 0; i < 8; ++i) { r[i] = a[i]; r[i + 8] = b[i]; }
  return r;
}

// One WMMA layer: A (LDS f16, 64 x K) * Bpack (global f16) + bias, ReLU -> C (LDS f16)
static __device__ __forceinline__ void gemm_relu(
    const _Float16* A, int Astr,
    const _Float16* Bw,
    const float* bias,
    _Float16* C, int Cstr,
    int Ntiles, int Ksteps, int wave, int lane)
{
  const int ln15 = lane & 15;
  const int koff = (lane < 16) ? 0 : 16;
#pragma unroll 1
  for (int tt = wave; tt < 4 * Ntiles; tt += 8) {
    const int mt = tt & 3;
    const int nt = tt >> 2;
    v8f acc = {};
    const _Float16* Arow = A + (mt * 16 + ln15) * Astr + koff;
    const _Float16* Bcol = Bw + (size_t)(nt * Ksteps) * 512 + lane * 16;
    // unroll-by-2: clause 4 global + 4 LDS loads, overlap WMMA with next loads
#pragma unroll 2
    for (int kt = 0; kt < Ksteps; ++kt) {
      v8h a0 = *(const v8h*)(Arow + kt * 32);
      v8h a1 = *(const v8h*)(Arow + kt * 32 + 8);
      v8h b0 = *(const v8h*)(Bcol + (size_t)kt * 512);
      v8h b1 = *(const v8h*)(Bcol + (size_t)kt * 512 + 8);
      acc = __builtin_amdgcn_wmma_f32_16x16x32_f16(false, cat16(a0, a1),
                                                   false, cat16(b0, b1),
                                                   (short)0, acc, false, false);
    }
    const int n   = nt * 16 + ln15;
    const float b = bias[n];
    const int m0  = mt * 16 + ((lane < 16) ? 0 : 8);
#pragma unroll
    for (int r = 0; r < 8; ++r) {
      float v = acc[r] + b;
      v = v > 0.0f ? v : 0.0f;
      C[(m0 + r) * Cstr + n] = (_Float16)v;
    }
  }
}

// Pack W[out_f][in_f] (f32) into WMMA B-operand blocks: per (kt,nt) 32x16 block,
// lane<16 holds (k=kt*32+i, n=nt*16+lane), lane>=16 holds (k=kt*32+16+i, ...),
// 16 contiguous halves per lane -> 32B/lane coalesced fetch in the GEMM.
__global__ void pack_weights(const float* __restrict__ W, _Float16* __restrict__ dst,
                             int out_f, int in_f, int Kp, int Np)
{
  int idx = blockIdx.x * blockDim.x + threadIdx.x;
  int total = Kp * Np;
  if (idx >= total) return;
  int n = idx / Kp;
  int k = idx - n * Kp;
  int kt = k >> 5, kr = k & 31;
  int nt = n >> 4, nr = n & 15;
  int lanep = nr + (kr & 16);
  int half  = kr & 15;
  int Ksteps = Kp >> 5;
  float v = (k < in_f && n < out_f) ? W[(size_t)n * in_f + k] : 0.0f;
  dst[(size_t)(nt * Ksteps + kt) * 512 + lanep * 16 + half] = (_Float16)v;
}

__global__ void pack_biases(const float* b1, const float* b2, const float* b3,
                            const float* b4, float* o)
{
  int i = blockIdx.x * blockDim.x + threadIdx.x;
  if (i < 272)      { o[i] = (i < 270) ? b1[i] : 0.0f; }
  else if (i < 368) { int j = i - 272; o[i] = (j < 90) ? b2[j] : 0.0f; }
  else if (i < 400) { int j = i - 368; o[i] = (j < 30) ? b3[j] : 0.0f; }
  else if (i < 416) { int j = i - 400; o[i] = (j < 10) ? b4[j] : 0.0f; }
}

__global__ __launch_bounds__(256)
void fused_stats_mlp(const float* __restrict__ t,
                     const _Float16* __restrict__ wpack,
                     const float* __restrict__ biases,
                     float* __restrict__ out, int B)
{
  extern __shared__ char smem_raw[];
  _Float16* feat = (_Float16*)smem_raw;                 // [64][424] f16
  _Float16* act1 = feat + TILE_M * FSTR;                // [64][296]
  _Float16* act2 = act1 + TILE_M * A1STR;               // [64][104]
  _Float16* act3 = act2 + TILE_M * A2STR;               // [64][40]
  float*  logits = (float*)(act3 + TILE_M * A3STR);     // [64][16]

  const int tid  = threadIdx.x;
  const int lane = tid & 31;
  const int wave = tid >> 5;
  const int img0 = blockIdx.x * TILE_M;

  // Zero K-pad columns so uninitialized LDS never feeds the WMMA (NaN*0 hazard).
#pragma unroll 1
  for (int i = tid; i < TILE_M * 24; i += 256)
    feat[(i / 24) * FSTR + 392 + (i % 24)] = (_Float16)0.0f;
#pragma unroll 1
  for (int i = tid; i < TILE_M * 16; i += 256)
    act1[(i / 16) * A1STR + 272 + (i % 16)] = (_Float16)0.0f;

  // axis=2 reductions (over columns -> indexed by row): *_2 feature slots
#pragma unroll 1
  for (int rt = tid; rt < TILE_M * 28; rt += 256) {
    int m = rt / 28, r = rt - m * 28;
    int img = img0 + m;
    if (img < B) {
      stat_task(t + (size_t)img * 784 + r * 28, 1, feat + m * FSTR,
                56 + r, 84 + r, 168 + r, 196 + r, 252 + r, 336 + r, 364 + r);
    }
  }
  // axis=1 reductions (over rows -> indexed by column): *_1 feature slots
#pragma unroll 1
  for (int ct = tid; ct < TILE_M * 28; ct += 256) {
    int m = ct / 28, c = ct - m * 28;
    int img = img0 + m;
    if (img < B) {
      stat_task(t + (size_t)img * 784 + c, 28, feat + m * FSTR,
                0 + c, 28 + c, 112 + c, 140 + c, 224 + c, 280 + c, 308 + c);
    }
  }
  __syncthreads();

  // L1: [64x416] x [416x272]
  gemm_relu(feat, FSTR, wpack + OFF_W1, biases + 0,   act1, A1STR, 17, 13, wave, lane);
  __syncthreads();
  // L2: [64x288] x [288x96]
  gemm_relu(act1, A1STR, wpack + OFF_W2, biases + 272, act2, A2STR, 6, 9, wave, lane);
  __syncthreads();
  // L3: [64x96] x [96x32]
  gemm_relu(act2, A2STR, wpack + OFF_W3, biases + 368, act3, A3STR, 2, 3, wave, lane);
  __syncthreads();

  // L4: [64x32] x [32x16] -> f32 logits (no ReLU)
  if (wave < 4) {
    const int ln15 = lane & 15;
    const int koff = (lane < 16) ? 0 : 16;
    const int mt = wave;
    const _Float16* Arow = act3 + (mt * 16 + ln15) * A3STR + koff;
    v8h a0 = *(const v8h*)(Arow);
    v8h a1 = *(const v8h*)(Arow + 8);
    const _Float16* Bp = wpack + OFF_W4 + lane * 16;
    v8h b0 = *(const v8h*)(Bp);
    v8h b1 = *(const v8h*)(Bp + 8);
    v8f acc = {};
    acc = __builtin_amdgcn_wmma_f32_16x16x32_f16(false, cat16(a0, a1),
                                                 false, cat16(b0, b1),
                                                 (short)0, acc, false, false);
    const float bb = biases[400 + ln15];
    const int m0 = mt * 16 + ((lane < 16) ? 0 : 8);
#pragma unroll
    for (int r = 0; r < 8; ++r)
      logits[(m0 + r) * 16 + ln15] = acc[r] + bb;
  }
  __syncthreads();

  // softmax over the 10 valid logits per row
  if (tid < TILE_M) {
    int img = img0 + tid;
    if (img < B) {
      float mx = logits[tid * 16];
#pragma unroll
      for (int j = 1; j < 10; ++j) mx = fmaxf(mx, logits[tid * 16 + j]);
      float e[10]; float s = 0.0f;
#pragma unroll
      for (int j = 0; j < 10; ++j) { e[j] = __expf(logits[tid * 16 + j] - mx); s += e[j]; }
      float inv = 1.0f / s;
#pragma unroll
      for (int j = 0; j < 10; ++j) out[(size_t)img * 10 + j] = e[j] * inv;
    }
  }
}

extern "C" void kernel_launch(void* const* d_in, const int* in_sizes, int n_in,
                              void* d_out, int out_size, void* d_ws, size_t ws_size,
                              hipStream_t stream) {
  const float* t  = (const float*)d_in[0];
  const float* W1 = (const float*)d_in[1];
  const float* b1 = (const float*)d_in[2];
  const float* W2 = (const float*)d_in[3];
  const float* b2 = (const float*)d_in[4];
  const float* W3 = (const float*)d_in[5];
  const float* b3 = (const float*)d_in[6];
  const float* W4 = (const float*)d_in[7];
  const float* b4 = (const float*)d_in[8];
  float* out = (float*)d_out;
  const int B = in_sizes[0] / 784;

  _Float16* wpack = (_Float16*)d_ws;
  float* biases = (float*)((char*)d_ws + BIAS_BYTE_OFF);

  // Pack weights into WMMA B-operand layout (stays L2-resident: ~290 KB total).
  pack_weights<<<(416 * 272 + 255) / 256, 256, 0, stream>>>(W1, wpack + OFF_W1, 270, 392, 416, 272);
  pack_weights<<<(288 *  96 + 255) / 256, 256, 0, stream>>>(W2, wpack + OFF_W2,  90, 270, 288,  96);
  pack_weights<<<( 96 *  32 + 255) / 256, 256, 0, stream>>>(W3, wpack + OFF_W3,  30,  90,  96,  32);
  pack_weights<<<( 32 *  16 + 255) / 256, 256, 0, stream>>>(W4, wpack + OFF_W4,  10,  30,  32,  16);
  pack_biases<<<2, 256, 0, stream>>>(b1, b2, b3, b4, biases);

  const size_t smem = (size_t)TILE_M * (FSTR + A1STR + A2STR + A3STR) * sizeof(_Float16)
                    + (size_t)TILE_M * 16 * sizeof(float);   // 114688 B < 320 KB/WGP
  const int nblocks = (B + TILE_M - 1) / TILE_M;
  fused_stats_mlp<<<nblocks, 256, smem, stream>>>(t, wpack, biases, out, B);
}
